// MultiInputLSTM_39161511805119
// MI455X (gfx1250) — matrix operand order repaired and checked by reading, not compiled, not run on hardware
//
#include <hip/hip_runtime.h>

// ---------------------------------------------------------------------------
// MultiInputLSTM on gfx1250 (MI455X): bf16 WMMA for all GEMMs, f32 accumulate.
// Round 3: pin software pipelining with sched_barrier so K+1 fragment loads
// stay in flight across K-step WMMAs; prefetch epilogue operands in k_gates;
// 4-wave k_attn with uniform epilogue loop.
// ---------------------------------------------------------------------------

#define IN_DIM 256
#define HDIM   512
#define BDIM   64
#define TDIM   512
#define NG     8
#define BH     (BDIM * HDIM)       // 32768
#define BT     (BDIM * TDIM)       // 32768

#if __has_builtin(__builtin_amdgcn_sched_barrier)
#define SCHED_FENCE() __builtin_amdgcn_sched_barrier(0)
#else
#define SCHED_FENCE()
#endif

typedef __attribute__((ext_vector_type(16))) __bf16 bf16x16;
typedef __attribute__((ext_vector_type(8)))  __bf16 bf16x8;
typedef __attribute__((ext_vector_type(8)))  float  f32x8;

union BFrag { bf16x16 v; bf16x8 h[2]; };

// A fragment: 16x32 bf16 tile, row-major source, row stride lda.
// ISA layout: lanes 0-15 hold rows M=0..15 with K = k0+{0..7} (v0..3) and
// k0+16+{0..7} (v4..7); lanes 16-31 hold the +8 K chunks.
__device__ __forceinline__ bf16x16 load_a_frag(const __bf16* __restrict__ A,
                                               int lda, int row0, int k0) {
  const int lane = threadIdx.x & 31;
  const int l = lane & 15, hi = lane >> 4;
  const __bf16* p = A + (size_t)(row0 + l) * lda + k0 + hi * 8;
  BFrag f;
  f.h[0] = *(const bf16x8*)(p);        // K chunk 1 (16B)
  f.h[1] = *(const bf16x8*)(p + 16);   // K chunk 2 (16B)
  return f.v;
}

// B fragment: 32x16 bf16 tile from a PRE-TRANSPOSED weight (Bt[n][k], row
// stride ldb). ISA layout: lane = column N, lanes 0-15 hold K=k0+{0..7} and
// k0+16+{0..7}; lanes 16-31 hold the +8 K chunks (same order as A).
__device__ __forceinline__ bf16x16 load_b_frag(const __bf16* __restrict__ Bt,
                                               int ldb, int n0, int k0) {
  const int lane = threadIdx.x & 31;
  const int l = lane & 15, hi = lane >> 4;
  const __bf16* p = Bt + (size_t)(n0 + l) * ldb + k0 + hi * 8;
  BFrag f;
  f.h[0] = *(const bf16x8*)(p);
  f.h[1] = *(const bf16x8*)(p + 16);
  return f.v;
}

__device__ __forceinline__ f32x8 wmma_bf16(bf16x16 a, bf16x16 b, f32x8 c) {
  return __builtin_amdgcn_wmma_f32_16x16x32_bf16(
      /*neg_a=*/false, a, /*neg_b=*/false, b,
      /*c_mod=*/(short)0, c, /*reuse_a=*/false, /*reuse_b=*/false);
}

__device__ __forceinline__ float sigmoidf_(float x) {
  return 1.0f / (1.0f + __expf(-x));
}

// ---------------------------------------------------------------------------
// one-time prep kernels
// ---------------------------------------------------------------------------
__global__ void k_cvt_bf16(const float* __restrict__ in,
                           __bf16* __restrict__ out, int n) {
  int i = blockIdx.x * blockDim.x + threadIdx.x;
  if (i < n) out[i] = (__bf16)in[i];
}

// [G][R][C] f32 -> [G][C][R] bf16 (so K becomes contiguous for B-fragments)
__global__ void k_transpose_bf16(const float* __restrict__ in,
                                 __bf16* __restrict__ out,
                                 int G, int R, int C) {
  long i = blockIdx.x * (long)blockDim.x + threadIdx.x;
  long total = (long)G * R * C;
  if (i >= total) return;
  int c = (int)(i % C);
  long t = i / C;
  int r = (int)(t % R);
  int g = (int)(t / R);
  out[((long)g * C + c) * R + r] = (__bf16)in[i];
}

// ---------------------------------------------------------------------------
// xproj[t][g][b][h] = X_g(b,t,:) . Wx[g](:,h) + bias[g][h]
// A = selected input as (B*T x 256) row-major bf16, B = WxT[g] (512x256).
// Wave tile: 16 rows x 64 cols; pinned K pipeline (8 steps of 32).
// ---------------------------------------------------------------------------
__global__ __launch_bounds__(256)
void k_xproj(const __bf16* __restrict__ Ybf, const __bf16* __restrict__ Pbf,
             const __bf16* __restrict__ Nbf, const __bf16* __restrict__ WxT,
             const float* __restrict__ bias, float* __restrict__ xproj) {
  const int g = blockIdx.z;
  const __bf16* A = (g == 1 || g == 5) ? Pbf
                  : (g == 2 || g == 6) ? Nbf : Ybf;
  const __bf16* Bt = WxT + (size_t)g * HDIM * IN_DIM;
  const float* bg = bias + g * HDIM;
  const int wave = threadIdx.x >> 5;
  const int lane = threadIdx.x & 31;
  const int l = lane & 15, hi = lane >> 4;
  const int row0 = blockIdx.x * 128 + wave * 16;
  const int col0 = blockIdx.y * 64;

  f32x8 acc[4] = {};
  bf16x16 acur = load_a_frag(A, IN_DIM, row0, 0);
  bf16x16 bcur[4];
#pragma unroll
  for (int nt = 0; nt < 4; ++nt)
    bcur[nt] = load_b_frag(Bt, IN_DIM, col0 + nt * 16, 0);

#pragma unroll
  for (int ks = 0; ks < IN_DIM / 32 - 1; ++ks) {
    const int kn = (ks + 1) * 32;
    bf16x16 anxt = load_a_frag(A, IN_DIM, row0, kn);
    bf16x16 bnxt[4];
#pragma unroll
    for (int nt = 0; nt < 4; ++nt)
      bnxt[nt] = load_b_frag(Bt, IN_DIM, col0 + nt * 16, kn);
    SCHED_FENCE();  // keep K+1 loads issued before K-step WMMAs
#pragma unroll
    for (int nt = 0; nt < 4; ++nt)
      acc[nt] = wmma_bf16(acur, bcur[nt], acc[nt]);
    acur = anxt;
#pragma unroll
    for (int nt = 0; nt < 4; ++nt) bcur[nt] = bnxt[nt];
  }
#pragma unroll
  for (int nt = 0; nt < 4; ++nt)
    acc[nt] = wmma_bf16(acur, bcur[nt], acc[nt]);

#pragma unroll
  for (int nt = 0; nt < 4; ++nt) {
    const int h = col0 + nt * 16 + l;
    const float bb = bg[h];
#pragma unroll
    for (int r = 0; r < 8; ++r) {
      const int row = row0 + hi * 8 + r;       // row = b*T + t
      const int b_ = row >> 9;                 // /512
      const int t_ = row & 511;
      xproj[(((size_t)t_ * NG + g) * BDIM + b_) * HDIM + h] = acc[nt][r] + bb;
    }
  }
}

// ---------------------------------------------------------------------------
// Phase A (per step): g = xproj_t + h @ Uh[g]; fused gate activations.
// 8 waves, wave == gate; 32 blocks cover H=512 in 16-col tiles; cross-gate
// exchange via LDS. xproj operands prefetched before the GEMM.
// ---------------------------------------------------------------------------
__global__ __launch_bounds__(256)
void k_gates(const __bf16* __restrict__ hbf, const __bf16* __restrict__ UhT,
             const float* __restrict__ xproj_t, __bf16* __restrict__ lbf,
             float* __restrict__ lf32, float* __restrict__ fbuf,
             float* __restrict__ obuf) {
  __shared__ float gbuf[NG][BDIM][16];  // 32 KB
  const int wave = threadIdx.x >> 5;    // gate index 0..7
  const int lane = threadIdx.x & 31;
  const int l = lane & 15, hi = lane >> 4;
  const int col0 = blockIdx.x * 16;
  const __bf16* Bt = UhT + (size_t)wave * HDIM * HDIM;

  // Prefetch epilogue operands: resolved while the GEMM runs.
  float xp[4][NG];
#pragma unroll
  for (int j = 0; j < 4; ++j) {
    const int idx = threadIdx.x + j * 256;
    const size_t off = (size_t)(idx >> 4) * HDIM + col0 + (idx & 15);
#pragma unroll
    for (int g = 0; g < NG; ++g) xp[j][g] = xproj_t[(size_t)g * BH + off];
  }

  f32x8 acc[4] = {};
  bf16x16 bcur = load_b_frag(Bt, HDIM, col0, 0);
  bf16x16 acur[4];
#pragma unroll
  for (int mt = 0; mt < 4; ++mt)
    acur[mt] = load_a_frag(hbf, HDIM, mt * 16, 0);

#pragma unroll
  for (int ks = 0; ks < HDIM / 32 - 1; ++ks) {
    const int kn = (ks + 1) * 32;
    bf16x16 bnxt = load_b_frag(Bt, HDIM, col0, kn);
    bf16x16 anxt[4];
#pragma unroll
    for (int mt = 0; mt < 4; ++mt)
      anxt[mt] = load_a_frag(hbf, HDIM, mt * 16, kn);
    SCHED_FENCE();  // keep K+1 loads issued before K-step WMMAs
#pragma unroll
    for (int mt = 0; mt < 4; ++mt)
      acc[mt] = wmma_bf16(acur[mt], bcur, acc[mt]);
    bcur = bnxt;
#pragma unroll
    for (int mt = 0; mt < 4; ++mt) acur[mt] = anxt[mt];
  }
#pragma unroll
  for (int mt = 0; mt < 4; ++mt)
    acc[mt] = wmma_bf16(acur[mt], bcur, acc[mt]);

#pragma unroll
  for (int mt = 0; mt < 4; ++mt)
#pragma unroll
    for (int r = 0; r < 8; ++r)
      gbuf[wave][mt * 16 + hi * 8 + r][l] = acc[mt][r];
  __syncthreads();

#pragma unroll
  for (int j = 0; j < 4; ++j) {
    const int idx = threadIdx.x + j * 256;
    const int b_ = idx >> 4;
    const int c_ = idx & 15;
    const int h = col0 + c_;
    const size_t off = (size_t)b_ * HDIM + h;
    float gv[NG];
#pragma unroll
    for (int g = 0; g < NG; ++g) gv[g] = gbuf[g][b_][c_] + xp[j][g];
    const float i_t = sigmoidf_(gv[0]);
    const float i_p = sigmoidf_(gv[1]);
    const float i_n = sigmoidf_(gv[2]);
    const float f_t = sigmoidf_(gv[3]);
    const float C_t = tanhf(gv[4]);
    const float C_p = tanhf(gv[5]);
    const float C_n = tanhf(gv[6]);
    const float o_t = sigmoidf_(gv[7]);
    const float l0 = C_t * i_t, l1 = C_p * i_p, l2 = C_n * i_n;
    lf32[off] = l0; lf32[BH + off] = l1; lf32[2 * BH + off] = l2;
    lbf[off] = (__bf16)l0; lbf[BH + off] = (__bf16)l1;
    lbf[2 * BH + off] = (__bf16)l2;
    fbuf[off] = f_t;
    obuf[off] = o_t;
  }
}

// ---------------------------------------------------------------------------
// Phase B (per step): u = tanh((l @ W_a)*c + b_a); softmax over the 3
// branches; c/h update. 4 waves: waves 0-2 run the branch GEMMs, all 4
// waves share the fused epilogue (uniform 8-iteration loop).
// ---------------------------------------------------------------------------
__global__ __launch_bounds__(128)
void k_attn(const __bf16* __restrict__ lbf, const float* __restrict__ lf32,
            const __bf16* __restrict__ WaT, const float* __restrict__ b_a,
            const float* __restrict__ fbuf, const float* __restrict__ obuf,
            float* __restrict__ cbuf, __bf16* __restrict__ hbf,
            float* __restrict__ out_hs, float* __restrict__ out_hT, int t) {
  __shared__ float ubuf[3][BDIM][16];  // 12 KB
  const int wave = threadIdx.x >> 5;   // branch s = 0..2 (wave 3 idles in GEMM)
  const int lane = threadIdx.x & 31;
  const int l = lane & 15, hi = lane >> 4;
  const int col0 = blockIdx.x * 16;

  if (wave < 3) {
    const __bf16* A = lbf + (size_t)wave * BH;
    f32x8 acc[4] = {};
    bf16x16 bcur = load_b_frag(WaT, HDIM, col0, 0);
    bf16x16 acur[4];
#pragma unroll
    for (int mt = 0; mt < 4; ++mt)
      acur[mt] = load_a_frag(A, HDIM, mt * 16, 0);

#pragma unroll
    for (int ks = 0; ks < HDIM / 32 - 1; ++ks) {
      const int kn = (ks + 1) * 32;
      bf16x16 bnxt = load_b_frag(WaT, HDIM, col0, kn);
      bf16x16 anxt[4];
#pragma unroll
      for (int mt = 0; mt < 4; ++mt)
        anxt[mt] = load_a_frag(A, HDIM, mt * 16, kn);
      SCHED_FENCE();  // keep K+1 loads issued before K-step WMMAs
#pragma unroll
      for (int mt = 0; mt < 4; ++mt)
        acc[mt] = wmma_bf16(acur[mt], bcur, acc[mt]);
      bcur = bnxt;
#pragma unroll
      for (int mt = 0; mt < 4; ++mt) acur[mt] = anxt[mt];
    }
#pragma unroll
    for (int mt = 0; mt < 4; ++mt)
      acc[mt] = wmma_bf16(acur[mt], bcur, acc[mt]);

#pragma unroll
    for (int mt = 0; mt < 4; ++mt)
#pragma unroll
      for (int r = 0; r < 8; ++r)
        ubuf[wave][mt * 16 + hi * 8 + r][l] = acc[mt][r];
  }
  __syncthreads();

#pragma unroll
  for (int j = 0; j < 8; ++j) {
    const int idx = threadIdx.x + j * 128;
    const int b_ = idx >> 4;
    const int c_ = idx & 15;
    const int h = col0 + c_;
    const size_t off = (size_t)b_ * HDIM + h;
    const float c_old = cbuf[off];
    const float ba = b_a[h];
    const float u0 = tanhf(ubuf[0][b_][c_] * c_old + ba);
    const float u1 = tanhf(ubuf[1][b_][c_] * c_old + ba);
    const float u2 = tanhf(ubuf[2][b_][c_] * c_old + ba);
    const float m = fmaxf(u0, fmaxf(u1, u2));
    const float e0 = __expf(u0 - m), e1 = __expf(u1 - m), e2 = __expf(u2 - m);
    const float inv = 1.0f / (e0 + e1 + e2);
    const float L =
        (e0 * lf32[off] + e1 * lf32[BH + off] + e2 * lf32[2 * BH + off]) * inv;
    const float c_new = fbuf[off] * c_old + L;
    const float h_new = obuf[off] * tanhf(c_new);
    cbuf[off] = c_new;
    hbf[off] = (__bf16)h_new;
    out_hs[((size_t)b_ * TDIM + t) * HDIM + h] = h_new;
    if (t == TDIM - 1) out_hT[off] = h_new;
  }
}

// ---------------------------------------------------------------------------
// host launcher
// ---------------------------------------------------------------------------
extern "C" void kernel_launch(void* const* d_in, const int* in_sizes, int n_in,
                              void* d_out, int out_size, void* d_ws,
                              size_t ws_size, hipStream_t stream) {
  (void)in_sizes; (void)n_in; (void)out_size; (void)ws_size;
  const float* Y  = (const float*)d_in[0];
  const float* P  = (const float*)d_in[1];
  const float* N  = (const float*)d_in[2];
  const float* Wx = (const float*)d_in[3];
  const float* Uh = (const float*)d_in[4];
  const float* bg = (const float*)d_in[5];
  const float* Wa = (const float*)d_in[6];
  const float* ba = (const float*)d_in[7];
  float* out = (float*)d_out;

  char* ws = (char*)d_ws;
  size_t off = 0;
  auto alloc = [&](size_t bytes) -> char* {
    char* p = ws + off;
    off += (bytes + 255) & ~(size_t)255;
    return p;
  };

  __bf16* Ybf  = (__bf16*)alloc((size_t)BT * IN_DIM * 2);
  __bf16* Pbf  = (__bf16*)alloc((size_t)BT * IN_DIM * 2);
  __bf16* Nbf  = (__bf16*)alloc((size_t)BT * IN_DIM * 2);
  __bf16* WxT  = (__bf16*)alloc((size_t)NG * IN_DIM * HDIM * 2);
  __bf16* UhT  = (__bf16*)alloc((size_t)NG * HDIM * HDIM * 2);
  __bf16* WaT  = (__bf16*)alloc((size_t)HDIM * HDIM * 2);
  float*  xproj = (float*)alloc((size_t)TDIM * NG * BH * 4);
  __bf16* lbf  = (__bf16*)alloc((size_t)3 * BH * 2);
  float*  lf32 = (float*)alloc((size_t)3 * BH * 4);
  float*  fbuf = (float*)alloc((size_t)BH * 4);
  float*  obuf = (float*)alloc((size_t)BH * 4);
  float*  cbuf = (float*)alloc((size_t)BH * 4);
  __bf16* hbf  = (__bf16*)alloc((size_t)BH * 2);

  // One-time conversions / transposes.
  const int nBT = BT * IN_DIM;
  k_cvt_bf16<<<(nBT + 255) / 256, 256, 0, stream>>>(Y, Ybf, nBT);
  k_cvt_bf16<<<(nBT + 255) / 256, 256, 0, stream>>>(P, Pbf, nBT);
  k_cvt_bf16<<<(nBT + 255) / 256, 256, 0, stream>>>(N, Nbf, nBT);
  {
    long n = (long)NG * IN_DIM * HDIM;
    k_transpose_bf16<<<(unsigned)((n + 255) / 256), 256, 0, stream>>>(
        Wx, WxT, NG, IN_DIM, HDIM);
  }
  {
    long n = (long)NG * HDIM * HDIM;
    k_transpose_bf16<<<(unsigned)((n + 255) / 256), 256, 0, stream>>>(
        Uh, UhT, NG, HDIM, HDIM);
  }
  {
    long n = (long)HDIM * HDIM;
    k_transpose_bf16<<<(unsigned)((n + 255) / 256), 256, 0, stream>>>(
        Wa, WaT, 1, HDIM, HDIM);
  }

  // h0 = 0, c0 = 0 (graph-capturable memsets).
  hipMemsetAsync(hbf, 0, (size_t)BH * 2, stream);
  hipMemsetAsync(cbuf, 0, (size_t)BH * 4, stream);

  // Hoisted input projections: [T][8][B][H].
  k_xproj<<<dim3(BT / 128, HDIM / 64, NG), 256, 0, stream>>>(
      Ybf, Pbf, Nbf, WxT, bg, xproj);

  // Sequential recurrence: stream ordering provides the dependencies.
  float* out_hT = out;            // [B][H]
  float* out_hs = out + BH;       // [B][T][H]
  for (int t = 0; t < TDIM; ++t) {
    k_gates<<<HDIM / 16, 256, 0, stream>>>(
        hbf, UhT, xproj + (size_t)t * NG * BH, lbf, lf32, fbuf, obuf);
    k_attn<<<HDIM / 16, 128, 0, stream>>>(
        lbf, lf32, WaT, ba, fbuf, obuf, cbuf, hbf, out_hs, out_hT, t);
  }
}
